// CSABlock_32624571580850
// MI455X (gfx1250) — compile-verified, matrix operand order checked by reading, not run on hardware
//
#include <hip/hip_runtime.h>
#include <hip/hip_bf16.h>

// ---------------------------------------------------------------------------
// CSWin-style block on gfx1250.
//  - all GEMM-like math via v_wmma_f32_16x16x32_bf16 (fp32 accumulate)
//  - whole activation pipeline carried in bf16 (residual stream stays fp32)
//  - GEMM tile staging via global_load_async_to_lds_b128 (ASYNCcnt path)
// ---------------------------------------------------------------------------

typedef unsigned short u16;
typedef __attribute__((ext_vector_type(8)))  float  v8f;
typedef __attribute__((ext_vector_type(16))) __bf16 v16bf;

struct alignas(16) U4 { unsigned x, y, z, w; };

union BFrag { v16bf v; U4 q[2]; };

__device__ __forceinline__ u16 f2bf(float f) {
    unsigned u = __float_as_uint(f);
    u += 0x7FFFu + ((u >> 16) & 1u);          // round-to-nearest-even
    return (u16)(u >> 16);
}
__device__ __forceinline__ float bf2f(u16 h) {
    return __uint_as_float(((unsigned)h) << 16);
}

// CDNA5 16-bit A/B operand layout (ISA 7.12.2):
//   lane = tid&31, row/col = lane&15, hi = lane>>4
//   halves 0..7  = K: hi*8 + 0..7      (contiguous, 16B)
//   halves 8..15 = K: 16 + hi*8 + 0..7 (contiguous, 16B)
__device__ __forceinline__ v16bf frag_ld(const u16* rowp, int hi) {
    BFrag f;
    f.q[0] = *(const U4*)(rowp + hi * 8);
    f.q[1] = *(const U4*)(rowp + 16 + hi * 8);
    return f.v;
}

__device__ __forceinline__ v8f wmma_bf16(v16bf a, v16bf b, v8f c) {
    return __builtin_amdgcn_wmma_f32_16x16x32_bf16(false, a, false, b, (short)0, c,
                                                   false, false);
}

__device__ __forceinline__ void lds_fence() {
    asm volatile("s_wait_dscnt 0x0" ::: "memory");
}

// Async 16B global -> LDS copy (GLOBAL_LOAD_ASYNC_TO_LDS_B128, ASYNCcnt).
// LDS byte offset = low 32 bits of the generic shared-pointer address
// (ISA 10.2: LDS aperture maps addr[31:0] to the LDS offset).
__device__ __forceinline__ void async_ld_b128(void* ldsp, const void* gaddr) {
    unsigned loff = (unsigned)(unsigned long long)ldsp;
    asm volatile("global_load_async_to_lds_b128 %0, %1, off"
                 :: "v"(loff), "v"(gaddr)
                 : "memory");
}
__device__ __forceinline__ void wait_async() {
    asm volatile("s_wait_asynccnt 0x0" ::: "memory");
}

// ---------------------------------------------------------------------------
// Prep kernels (tiny, run once per launch)
// ---------------------------------------------------------------------------

__global__ void prep_bn_kernel(const float* __restrict__ cb, const float* __restrict__ g,
                               const float* __restrict__ bb, const float* __restrict__ m,
                               const float* __restrict__ v, float* __restrict__ S,
                               float* __restrict__ T) {
    int c = blockIdx.x * 256 + threadIdx.x;
    if (c >= 256) return;
    float rs = rsqrtf(v[c] + 1e-5f);
    S[c] = rs * g[c];
    T[c] = (cb[c] - m[c]) * rs * g[c] + bb[c];
}

// conv_w (co,ci,3,3) fp32 -> wT[(ky*3+kx)][co][ci] bf16 (ci contiguous)
__global__ void prep_convw_kernel(const float* __restrict__ w, u16* __restrict__ wT) {
    int i = blockIdx.x * 256 + threadIdx.x;       // 9*256*256
    if (i >= 9 * 256 * 256) return;
    int ci = i & 255;
    int r  = i >> 8;
    int co = r & 255;
    int kk = r >> 8;
    wT[i] = f2bf(w[(co * 256 + ci) * 9 + kk]);
}

// src[K][N] fp32 -> dst[N][K] bf16
__global__ void transpose_bf16_kernel(const float* __restrict__ src, u16* __restrict__ dst,
                                      int K, int N) {
    long i = (long)blockIdx.x * 256 + threadIdx.x;
    if (i >= (long)N * K) return;
    int  k = (int)(i % K);
    long n = i / K;
    dst[n * K + k] = f2bf(src[k * (long)N + n]);
}

// ---------------------------------------------------------------------------
// 3x3 conv (implicit GEMM, 9 shifted WMMA passes) + folded BN.
// Output layout (B, L=4096, C=256) fp32 (feeds LN0 which produces fp32 t).
// ---------------------------------------------------------------------------
__global__ __launch_bounds__(256) void conv_kernel(const float* __restrict__ x1,
                                                   const float* __restrict__ x2,
                                                   const u16* __restrict__ wT,
                                                   const float* __restrict__ bnS,
                                                   const float* __restrict__ bnT,
                                                   float* __restrict__ outp) {
    __shared__ __align__(16) u16 sA[16 * 32];     // 16 pixels x 32 in-ch, bf16

    const int tid  = threadIdx.x;
    const int P    = blockIdx.x * 16;             // base pixel (16 x's in one row)
    const int b    = P >> 12;
    const int rem  = P & 4095;
    const int y    = rem >> 6;
    const int x0   = rem & 63;
    const int wv   = tid >> 5;
    const int lane = tid & 31;
    const int col  = lane & 15;
    const int hi   = lane >> 4;
    const int cobase = blockIdx.y * 128 + wv * 16;

    v8f acc;
#pragma unroll
    for (int r = 0; r < 8; ++r) acc[r] = 0.f;

    for (int ky = 0; ky < 3; ++ky) {
        const int yy = y + ky - 1;
        const bool yok = (yy >= 0) && (yy < 64);  // uniform across the block
        if (!yok) continue;
        for (int kx = 0; kx < 3; ++kx) {
            for (int cic = 0; cic < 8; ++cic) {
                __syncthreads();
                for (int i = tid; i < 512; i += 256) {
                    int p  = i >> 5;
                    int cl = i & 31;
                    int c  = (cic << 5) + cl;
                    int xx = x0 + p + kx - 1;
                    float v = 0.f;
                    if (xx >= 0 && xx < 64) {
                        const float* src = (c < 128) ? x1 : x2;
                        v = src[(((long)b * 128 + (c & 127)) * 64 + yy) * 64 + xx];
                    }
                    sA[p * 32 + cl] = f2bf(v);
                }
                __syncthreads();
                v16bf af = frag_ld(&sA[col * 32], hi);
                const u16* brow =
                    wT + ((long)((ky * 3 + kx) * 256 + cobase + col)) * 256 + cic * 32;
                v16bf bfr = frag_ld(brow, hi);
                acc = wmma_bf16(af, bfr, acc);
            }
        }
    }

    const int c = cobase + col;
    const float s  = bnS[c];
    const float t0 = bnT[c];
#pragma unroll
    for (int r = 0; r < 8; ++r) {
        int mloc = hi * 8 + r;
        int l    = y * 64 + x0 + mloc;
        outp[((long)b * 4096 + l) * 256 + c] = acc[r] * s + t0;
    }
}

// ---------------------------------------------------------------------------
// LayerNorm over C=256, one wave per row. OUT_BF16 chooses output precision:
// ln0 feeds the fp32 residual stream; n1/n2 feed bf16 GEMM A-operands.
// ---------------------------------------------------------------------------
template <int OUT_BF16>
__global__ __launch_bounds__(256) void ln_kernel(const float* __restrict__ src,
                                                 const float* __restrict__ g,
                                                 const float* __restrict__ bta,
                                                 void* __restrict__ dstv) {
    const int wv   = threadIdx.x >> 5;
    const int lane = threadIdx.x & 31;
    const long row = (long)blockIdx.x * 8 + wv;
    const float* rp = src + row * 256;

    float v[8];
    float sum = 0.f;
#pragma unroll
    for (int j = 0; j < 8; ++j) { v[j] = rp[lane + j * 32]; sum += v[j]; }
#pragma unroll
    for (int off = 16; off; off >>= 1) sum += __shfl_xor(sum, off, 32);
    const float mu = sum * (1.f / 256.f);

    float vs = 0.f;
#pragma unroll
    for (int j = 0; j < 8; ++j) { float d = v[j] - mu; vs += d * d; }
#pragma unroll
    for (int off = 16; off; off >>= 1) vs += __shfl_xor(vs, off, 32);
    const float inv = rsqrtf(vs * (1.f / 256.f) + 1e-5f);

#pragma unroll
    for (int j = 0; j < 8; ++j) {
        int c = lane + j * 32;
        float o = (v[j] - mu) * inv * g[c] + bta[c];
        if (OUT_BF16) ((u16*)dstv)[row * 256 + c] = f2bf(o);
        else          ((float*)dstv)[row * 256 + c] = o;
    }
}

// ---------------------------------------------------------------------------
// bf16-WMMA GEMM: out[M,N] = act(A[M,K] @ Bt[N,K]^T + bias) (+res)
// A and Bt are bf16; tiles staged through LDS with async b128 copies.
// BM=BN=128, BK=32; 8 waves, each 32x64.
// ---------------------------------------------------------------------------
template <int ACT, int HAS_BIAS, int HAS_RES, int OUT_BF16>
__global__ __launch_bounds__(256) void gemm_kernel(const u16* __restrict__ A,
                                                   const u16* __restrict__ Bt,
                                                   const float* __restrict__ bias,
                                                   const float* __restrict__ res,
                                                   void* __restrict__ outv,
                                                   int M, int N, int K) {
    __shared__ __align__(16) u16 sA[128 * 32];
    __shared__ __align__(16) u16 sB[128 * 32];

    const int tid  = threadIdx.x;
    const int mblk = blockIdx.y * 128;
    const int nblk = blockIdx.x * 128;
    const int wv   = tid >> 5;
    const int lane = tid & 31;
    const int col  = lane & 15;
    const int hi   = lane >> 4;
    const int wr   = wv & 3;       // 4 wave-rows
    const int wc   = wv >> 2;      // 2 wave-cols
    const int mW   = wr * 32;
    const int nW   = wc * 64;

    v8f acc[2][4];
#pragma unroll
    for (int i = 0; i < 2; ++i)
#pragma unroll
        for (int j = 0; j < 4; ++j)
#pragma unroll
            for (int r = 0; r < 8; ++r) acc[i][j][r] = 0.f;

    const int row = tid >> 1;           // 0..127
    const int seg = (tid & 1) * 16;     // 0 or 16 halves

    for (int kb = 0; kb < K; kb += 32) {
        __syncthreads();                // previous tile fully consumed
        {   // A tile: bf16 [M][K], 32B per thread via 2 async b128s
            const u16* src = A + (long)(mblk + row) * K + kb + seg;
            u16* dst = &sA[row * 32 + seg];
            async_ld_b128(dst, src);
            async_ld_b128(dst + 8, src + 8);
        }
        {   // B tile: bf16 [N][K]
            const u16* src = Bt + (long)(nblk + row) * K + kb + seg;
            u16* dst = &sB[row * 32 + seg];
            async_ld_b128(dst, src);
            async_ld_b128(dst + 8, src + 8);
        }
        wait_async();                   // this wave's async copies landed
        __syncthreads();                // everyone's copies landed

        v16bf af[2], bfr[4];
#pragma unroll
        for (int i = 0; i < 2; ++i) af[i]  = frag_ld(&sA[(mW + i * 16 + col) * 32], hi);
#pragma unroll
        for (int j = 0; j < 4; ++j) bfr[j] = frag_ld(&sB[(nW + j * 16 + col) * 32], hi);
#pragma unroll
        for (int i = 0; i < 2; ++i)
#pragma unroll
            for (int j = 0; j < 4; ++j) acc[i][j] = wmma_bf16(af[i], bfr[j], acc[i][j]);
    }

#pragma unroll
    for (int i = 0; i < 2; ++i)
#pragma unroll
        for (int j = 0; j < 4; ++j) {
            int n = nblk + nW + j * 16 + col;
            float bv = HAS_BIAS ? bias[n] : 0.f;
#pragma unroll
            for (int r = 0; r < 8; ++r) {
                int m = mblk + mW + i * 16 + hi * 8 + r;
                float val = acc[i][j][r] + bv;
                if (ACT == 1) val = 0.5f * val * (1.f + erff(val * 0.70710678118f));
                if (HAS_RES) val += res[(long)m * N + n];
                if (OUT_BF16) ((u16*)outv)[(long)m * N + n] = f2bf(val);
                else          ((float*)outv)[(long)m * N + n] = val;
            }
        }
}

// ---------------------------------------------------------------------------
// LePE window attention. One workgroup (2 waves) per (window, head).
// S=128 tokens, hd=32. QK^T and PV via WMMA; softmax + depthwise-3x3 LePE
// epilogue in VALU. qkv layout (B, L, 3, C) bf16; writes att (B, L, 256) bf16.
// ---------------------------------------------------------------------------
__global__ __launch_bounds__(64) void attn_kernel(const u16* __restrict__ qkv,
                                                  u16* __restrict__ att,
                                                  const float* __restrict__ lepe_w,
                                                  const float* __restrict__ lepe_b,
                                                  int Hs, int Ws, int chan_off) {
    __shared__ __align__(16) u16  sQ[128 * 32];       // [t][d], q * scale
    __shared__ __align__(16) u16  sK[128 * 32];       // [t][d]
    __shared__ __align__(16) u16  sV[32 * 128];       // [d][t] (transposed)
    __shared__ __align__(16) float sS[2][16 * 128];   // per-wave score rows
    __shared__ __align__(16) u16  sP[2][16 * 128];    // per-wave softmax bf16

    const int tid  = threadIdx.x;
    const int wh   = blockIdx.x;
    const int widx = wh >> 2;
    const int head = wh & 3;
    const int b    = widx >> 5;
    const int win  = widx & 31;
    const int nwx  = 64 / Ws;
    const int wy   = win / nwx;
    const int wx   = win % nwx;

    const float scale = 0.17677669529663687f;   // 32^-0.5

    for (int i = tid; i < 128 * 32; i += 64) {
        int t = i >> 5, d = i & 31;
        int hs = t / Ws, wsc = t % Ws;
        int yy = wy * Hs + hs, xx = wx * Ws + wsc;
        int l  = yy * 64 + xx;
        long base = ((long)(b * 4096 + l)) * 768 + chan_off + head * 32 + d;
        sQ[t * 32 + d]  = f2bf(bf2f(qkv[base]) * scale);
        sK[t * 32 + d]  = qkv[base + 256];
        sV[d * 128 + t] = qkv[base + 512];
    }
    __syncthreads();

    const int wv   = tid >> 5;
    const int lane = tid & 31;
    const int col  = lane & 15;
    const int hi   = lane >> 4;
    float* myS = &sS[wv][0];
    u16*   myP = &sP[wv][0];

    for (int mt = wv * 4; mt < wv * 4 + 4; ++mt) {
        const int mbase = mt * 16;

        // ---- scores = (q*scale) . k^T : 8 n-tiles, K=32 ----
        v16bf aq = frag_ld(&sQ[(mbase + col) * 32], hi);
#pragma unroll
        for (int nt = 0; nt < 8; ++nt) {
            v16bf bk = frag_ld(&sK[(nt * 16 + col) * 32], hi);
            v8f s;
#pragma unroll
            for (int r = 0; r < 8; ++r) s[r] = 0.f;
            s = wmma_bf16(aq, bk, s);
#pragma unroll
            for (int r = 0; r < 8; ++r)
                myS[(hi * 8 + r) * 128 + nt * 16 + col] = s[r];
        }
        lds_fence();

        // ---- softmax rows (lanes 0..15, one row each) -> bf16 sP ----
        if (lane < 16) {
            float* rp = &myS[lane * 128];
            float mx = -1e30f;
            for (int j = 0; j < 128; ++j) mx = fmaxf(mx, rp[j]);
            float sm = 0.f;
            for (int j = 0; j < 128; ++j) { float e = __expf(rp[j] - mx); rp[j] = e; sm += e; }
            float inv = 1.f / sm;
            u16* pp = &myP[lane * 128];
            for (int j = 0; j < 128; ++j) pp[j] = f2bf(rp[j] * inv);
        }
        lds_fence();

        // ---- o = P @ V (+LePE) : 2 d-tiles, K=128 (4 steps) ----
#pragma unroll
        for (int nt2 = 0; nt2 < 2; ++nt2) {
            v8f o;
#pragma unroll
            for (int r = 0; r < 8; ++r) o[r] = 0.f;
#pragma unroll
            for (int kk = 0; kk < 4; ++kk) {
                v16bf ap = frag_ld(&myP[col * 128 + kk * 32], hi);
                v16bf bv = frag_ld(&sV[(nt2 * 16 + col) * 128 + kk * 32], hi);
                o = wmma_bf16(ap, bv, o);
            }
            const int d     = nt2 * 16 + col;
            const int dchan = head * 32 + d;
            float wgt[9];
#pragma unroll
            for (int kk = 0; kk < 9; ++kk) wgt[kk] = lepe_w[dchan * 9 + kk];
            const float lbias = lepe_b[dchan];
#pragma unroll
            for (int r = 0; r < 8; ++r) {
                int s   = mbase + hi * 8 + r;
                int hs  = s / Ws, wsc = s % Ws;
                float lp = lbias;
#pragma unroll
                for (int ky = 0; ky < 3; ++ky) {
                    int h2 = hs + ky - 1;
                    if (h2 < 0 || h2 >= Hs) continue;
#pragma unroll
                    for (int kx = 0; kx < 3; ++kx) {
                        int w2 = wsc + kx - 1;
                        if (w2 < 0 || w2 >= Ws) continue;
                        lp += wgt[ky * 3 + kx] * bf2f(sV[d * 128 + h2 * Ws + w2]);
                    }
                }
                int yy = wy * Hs + hs, xx = wx * Ws + wsc;
                int l  = yy * 64 + xx;
                att[((long)(b * 4096 + l)) * 256 + chan_off + dchan] = f2bf(o[r] + lp);
            }
        }
        lds_fence();   // before next m-tile reuses myS/myP
    }
}

// ---------------------------------------------------------------------------
// (B,L,C) fp32 -> two NCHW outputs concatenated.
// ---------------------------------------------------------------------------
__global__ void out_kernel(const float* __restrict__ t, float* __restrict__ o) {
    long i = (long)blockIdx.x * 256 + threadIdx.x;   // 8*256*4096
    int  p = (int)(i & 4095);
    long r = i >> 12;
    int  c = (int)(r & 255);
    int  b = (int)(r >> 8);
    float v = t[((long)b * 4096 + p) * 256 + c];
    long dst = (c < 128) ? (((long)b * 128 + c) * 4096 + p)
                         : (4194304L + ((long)b * 128 + (c - 128)) * 4096 + p);
    o[dst] = v;
}

// ---------------------------------------------------------------------------
extern "C" void kernel_launch(void* const* d_in, const int* in_sizes, int n_in,
                              void* d_out, int out_size, void* d_ws, size_t ws_size,
                              hipStream_t stream) {
    const float* x1     = (const float*)d_in[0];
    const float* x2     = (const float*)d_in[1];
    const float* conv_w = (const float*)d_in[2];
    const float* conv_b = (const float*)d_in[3];
    const float* bn_g   = (const float*)d_in[4];
    const float* bn_b   = (const float*)d_in[5];
    const float* bn_m   = (const float*)d_in[6];
    const float* bn_v   = (const float*)d_in[7];
    const float* ln0_g  = (const float*)d_in[8];
    const float* ln0_b  = (const float*)d_in[9];
    const float* n1_g   = (const float*)d_in[10];
    const float* n1_b   = (const float*)d_in[11];
    const float* qkv_w  = (const float*)d_in[12];
    const float* lepe_w = (const float*)d_in[13];
    const float* lepe_b = (const float*)d_in[14];
    const float* proj_w = (const float*)d_in[15];
    const float* proj_b = (const float*)d_in[16];
    const float* n2_g   = (const float*)d_in[17];
    const float* n2_b   = (const float*)d_in[18];
    const float* fc1_w  = (const float*)d_in[19];
    const float* fc1_b  = (const float*)d_in[20];
    const float* fc2_w  = (const float*)d_in[21];
    const float* fc2_b  = (const float*)d_in[22];

    const long M = 32768;   // B * H * W

    char*  ws  = (char*)d_ws;
    size_t off = 0;
    auto alloc = [&](size_t bytes) -> char* {
        char* p = ws + off;
        off = (off + bytes + 255) & ~(size_t)255;
        return p;
    };
    float* t     = (float*)alloc(M * 256 * 4);     // persistent fp32 residual
    u16*   img   = (u16*)alloc(M * 256 * 2);       // bf16 LN outputs (img / y)
    u16*   qkvh  = (u16*)alloc(M * 1024 * 2);      // bf16 qkv (768) then hidden (1024)
    char*  tmp   = alloc(M * 256 * 4);             // fp32 conv temp / bf16 att
    float* convt = (float*)tmp;
    u16*   attb  = (u16*)tmp;
    float* bnS   = (float*)alloc(1024);
    float* bnT   = (float*)alloc(1024);
    u16* wTconv = (u16*)alloc((size_t)9 * 256 * 256 * 2);
    u16* qkvT   = (u16*)alloc((size_t)2 * 256 * 768 * 2);
    u16* projT  = (u16*)alloc((size_t)2 * 256 * 256 * 2);
    u16* fc1T   = (u16*)alloc((size_t)2 * 256 * 1024 * 2);
    u16* fc2T   = (u16*)alloc((size_t)2 * 1024 * 256 * 2);

    // ---- weight prep ----
    prep_bn_kernel<<<1, 256, 0, stream>>>(conv_b, bn_g, bn_b, bn_m, bn_v, bnS, bnT);
    prep_convw_kernel<<<(9 * 256 * 256) / 256, 256, 0, stream>>>(conv_w, wTconv);
    for (int i = 0; i < 2; ++i) {
        transpose_bf16_kernel<<<(256 * 768 + 255) / 256, 256, 0, stream>>>(
            qkv_w + (long)i * 256 * 768, qkvT + (long)i * 256 * 768, 256, 768);
        transpose_bf16_kernel<<<(256 * 256 + 255) / 256, 256, 0, stream>>>(
            proj_w + (long)i * 256 * 256, projT + (long)i * 256 * 256, 256, 256);
        transpose_bf16_kernel<<<(256 * 1024 + 255) / 256, 256, 0, stream>>>(
            fc1_w + (long)i * 256 * 1024, fc1T + (long)i * 256 * 1024, 256, 1024);
        transpose_bf16_kernel<<<(1024 * 256 + 255) / 256, 256, 0, stream>>>(
            fc2_w + (long)i * 1024 * 256, fc2T + (long)i * 1024 * 256, 1024, 256);
    }

    // ---- conv + BN -> (B,L,C) fp32, then LN0 -> fp32 residual stream t ----
    conv_kernel<<<dim3(2048, 2), 256, 0, stream>>>(x1, x2, wTconv, bnS, bnT, convt);
    ln_kernel<0><<<4096, 256, 0, stream>>>(convt, ln0_g, ln0_b, t);

    for (int i = 0; i < 2; ++i) {
        // img = LN(t) -> bf16
        ln_kernel<1><<<4096, 256, 0, stream>>>(t, n1_g + i * 256, n1_b + i * 256, img);
        // qkv = img @ Wqkv -> bf16
        gemm_kernel<0, 0, 0, 1><<<dim3(6, 256), 256, 0, stream>>>(
            img, qkvT + (long)i * 256 * 768, nullptr, nullptr, qkvh, (int)M, 768, 256);
        // two cross-shaped stripe branches (64x2 and 2x64 windows) -> bf16 att
        attn_kernel<<<1024, 64, 0, stream>>>(qkvh, attb,
                                             lepe_w + (long)(i * 2 + 0) * 128 * 9,
                                             lepe_b + (long)(i * 2 + 0) * 128, 64, 2, 0);
        attn_kernel<<<1024, 64, 0, stream>>>(qkvh, attb,
                                             lepe_w + (long)(i * 2 + 1) * 128 * 9,
                                             lepe_b + (long)(i * 2 + 1) * 128, 2, 64, 128);
        // t += att @ Wproj + b   (in-place residual, fp32 out)
        gemm_kernel<0, 1, 1, 0><<<dim3(2, 256), 256, 0, stream>>>(
            attb, projT + (long)i * 256 * 256, proj_b + i * 256, t, t, (int)M, 256, 256);
        // y = LN(t) bf16; h = gelu(y @ W1 + b1) bf16; t += h @ W2 + b2
        ln_kernel<1><<<4096, 256, 0, stream>>>(t, n2_g + i * 256, n2_b + i * 256, img);
        gemm_kernel<1, 1, 0, 1><<<dim3(8, 256), 256, 0, stream>>>(
            img, fc1T + (long)i * 256 * 1024, fc1_b + i * 1024, nullptr, qkvh,
            (int)M, 1024, 256);
        gemm_kernel<0, 1, 1, 0><<<dim3(2, 256), 256, 0, stream>>>(
            qkvh, fc2T + (long)i * 1024 * 256, fc2_b + i * 256, t, t, (int)M, 256, 1024);
    }

    out_kernel<<<32768, 256, 0, stream>>>(t, (float*)d_out);
}
